// MultiHeadAttention_87067577024711
// MI455X (gfx1250) — compile-verified, matrix operand order checked by reading, not run on hardware
//
#include <hip/hip_runtime.h>
#include <hip/hip_bf16.h>

// ---------------------------------------------------------------------------
// MHA forward for MI455X (gfx1250), bf16 WMMA path, fp32 accumulation.
// Pipeline: fp32->bf16 convert -> 3x projection GEMM (head-split output)
//           -> flash attention (causal, online softmax)
//           -> output projection GEMM (fp32 out).
// WMMA fragments are pairs of b128 loads: Q's from global, the rest from LDS tiles that
// are staged in fragment-native layout. Tile staging uses the gfx1250 async
// LDS path (GLOBAL_LOAD_ASYNC_TO_LDS_B128 + s_wait_asynccnt) when the
// toolchain exposes the builtins, else synchronous b128 copies.
// Workspace layout (needs 64 MB):
//   [ 0MB) q_bf16   8MB   [ 8MB) k_bf16  8MB   [16MB) v_bf16  8MB
//   [24MB) Wq_bf16  2MB   [26MB) Wk 2MB  [28MB) Wv 2MB  [30MB) Wo 2MB
//   [32MB) Qh 8MB   [40MB) Kh 8MB  [48MB) Vh 8MB  [56MB) attn_cat 8MB
// ---------------------------------------------------------------------------

typedef __bf16 bf16;
typedef __attribute__((ext_vector_type(16))) __bf16 v16bf;
typedef __attribute__((ext_vector_type(8)))  float  v8f;
typedef int v4i __attribute__((vector_size(16)));   // GCC-vector int4

constexpr int kD    = 1024;     // model dim
constexpr int kH    = 16;       // heads
constexpr int kDH   = 64;       // head dim
constexpr int kB    = 2;        // batch
constexpr int kS    = 2048;     // sequence
constexpr int kNTok = kB * kS;  // 4096 tokens

// ---- async global->LDS staging (gfx1250 ASYNCcnt path), guarded -----------
#if defined(__has_builtin)
#if __has_builtin(__builtin_amdgcn_global_load_async_to_lds_b128) && \
    __has_builtin(__builtin_amdgcn_s_wait_asynccnt)
#define USE_ASYNC_LDS 1
#endif
#endif
#ifndef USE_ASYNC_LDS
#define USE_ASYNC_LDS 0
#endif

__device__ inline void stage_b128(const bf16* gsrc, bf16* ldst) {
#if USE_ASYNC_LDS
  // Signature (from toolchain diagnostic): (global v4i*, lds v4i*, int, int)
  __builtin_amdgcn_global_load_async_to_lds_b128(
      (__attribute__((address_space(1))) v4i*)const_cast<bf16*>(gsrc),
      (__attribute__((address_space(3))) v4i*)ldst,
      /*offset=*/0, /*cpol=*/0);
#else
  *reinterpret_cast<uint4*>(ldst) = *reinterpret_cast<const uint4*>(gsrc);
#endif
}
__device__ inline void stage_join() {
#if USE_ASYNC_LDS
  __builtin_amdgcn_s_wait_asynccnt(0);
#endif
}

// round-to-nearest-even fp32 -> bf16 (pure bit ops)
__device__ inline bf16 to_bf16(float x) {
  unsigned u = __float_as_uint(x);
  unsigned r = (u + 0x7FFFu + ((u >> 16) & 1u)) >> 16;
  unsigned short h = (unsigned short)r;
  return __builtin_bit_cast(bf16, h);
}

__device__ inline v8f wmma_bf16(v16bf a, v16bf b, v8f c) {
  // D = A(16x32 bf16) * B(32x16 bf16) + C(16x16 f32)
  return __builtin_amdgcn_wmma_f32_16x16x32_bf16(
      /*neg_a=*/false, a, /*neg_b=*/false, b,
      /*c_mod=*/(short)0, c, /*reuse_a=*/false, /*reuse_b=*/false);
}

// Fragment load from a row-major tile where each lane's 16 elements are two
// contiguous 8-element (16B) runs -> two ds_load_b128.
// Used for A (lane -> matrix row, elems -> k) and for B from tiles stored as
// [n][k] (lane -> matrix col, elems -> k).  ISA 7.12.2 half-lane K split:
// lanes 0-15: K 0-7 / 16-23; lanes 16-31: K 8-15 / 24-31.
template <int LD>
__device__ inline v16bf load_frag(const bf16* p) {
  const int lane = threadIdx.x & 31;
  const int row  = lane & 15;
  const int kb   = (lane >> 4) * 8;
  struct alignas(16) P2 { uint4 lo, hi; } t;
  t.lo = *reinterpret_cast<const uint4*>(p + row * LD + kb);
  t.hi = *reinterpret_cast<const uint4*>(p + row * LD + kb + 16);
  return __builtin_bit_cast(v16bf, t);
}

struct B16x16 { bf16 v[16]; };   // 32B staging helper

// ---------------------------------------------------------------------------
__global__ void cvt_f32_bf16_kernel(const float* __restrict__ src,
                                    bf16* __restrict__ dst, int n) {
  int i = blockIdx.x * blockDim.x + threadIdx.x;
  const int stride = gridDim.x * blockDim.x;
  for (; i < n; i += stride) dst[i] = to_bf16(src[i]);
}

// ---------------------------------------------------------------------------
// Y[m,n] = sum_k X[m,k] * W[n,k] + bias[n]      (Y = X @ W^T + b)
// mode 0: write bf16 head-split layout [H, B, S, DH]
// mode 1: write fp32 [NTok, D] (final output)
// Block: 128 threads (4 waves). Tile: 64(M) x 128(N), K-step 32.
// W rows are staged UNTRANSPOSED into sBt[n][k]; the B-fragment gather over
// that layout is the same two-b128 pattern as the A-fragment.
// ---------------------------------------------------------------------------
constexpr int GBM = 64, GBN = 128, GBK = 32;
constexpr int PAD = 16;   // bf16 elems; keeps row stride a 16B multiple

__global__ __launch_bounds__(128) void gemm_bf16_kernel(
    const bf16* __restrict__ X, const bf16* __restrict__ W,
    const float* __restrict__ bias,
    bf16* __restrict__ outHead, float* __restrict__ outF32, int mode) {
  __shared__ __align__(16) bf16 sA[GBM][GBK + PAD];    // X tile  [m][k]
  __shared__ __align__(16) bf16 sBt[GBN][GBK + PAD];   // W tile  [n][k]

  const int tid  = threadIdx.x;
  const int wave = tid >> 5;
  const int lane = tid & 31;
  const int m0   = blockIdx.x * GBM;
  const int n0   = blockIdx.y * GBN;

  v8f acc[8] = {};

  for (int kk = 0; kk < kD; kk += GBK) {
    // Stage A (64x32): 2 x 16B per thread, async global -> LDS.
#pragma unroll
    for (int c = 0; c < 2; ++c) {
      const int chunk = tid * 2 + c;       // 0..255
      const int row   = chunk >> 2;        // 0..63
      const int colc  = (chunk & 3) * 8;   // 0,8,16,24
      stage_b128(X + (size_t)(m0 + row) * kD + kk + colc, &sA[row][colc]);
    }
    // Stage B (128x32): straight row copies of W, 4 x 16B per thread.
#pragma unroll
    for (int c = 0; c < 4; ++c) {
      const int chunk = tid * 4 + c;       // 0..511
      const int n     = chunk >> 2;        // 0..127
      const int colc  = (chunk & 3) * 8;   // 0,8,16,24
      stage_b128(W + (size_t)(n0 + n) * kD + kk + colc, &sBt[n][colc]);
    }
    if (kk + GBK < kD) {   // next K-tile hint -> global_prefetch_b8
      __builtin_prefetch(X + (size_t)(m0 + (tid >> 1)) * kD + kk + GBK, 0, 3);
      __builtin_prefetch(W + (size_t)(n0 + tid) * kD + kk + GBK, 0, 3);
    }
    stage_join();
    __syncthreads();

    // Pre-load all fragments, then an uninterrupted WMMA burst.
    const v16bf a = load_frag<GBK + PAD>(&sA[wave * 16][0]);
    v16bf bfr[8];
#pragma unroll
    for (int t = 0; t < 8; ++t)
      bfr[t] = load_frag<GBK + PAD>(&sBt[t * 16][0]);
#pragma unroll
    for (int t = 0; t < 8; ++t)
      acc[t] = wmma_bf16(a, bfr[t], acc[t]);
    __syncthreads();
  }

  // Epilogue. C layout: lane l, elem e -> row (l>>4)*8+e, col l&15.
  const int col   = lane & 15;
  const int rbase = (lane >> 4) * 8;
#pragma unroll
  for (int t = 0; t < 8; ++t) {
    const int n  = n0 + t * 16 + col;
    const float bn = bias[n];
#pragma unroll
    for (int e = 0; e < 8; ++e) {
      const int m = m0 + wave * 16 + rbase + e;
      const float y = acc[t][e] + bn;
      if (mode == 0) {
        const int h = n >> 6, hc = n & 63;            // head / col in head
        const int bb = m >> 11, ss = m & (kS - 1);    // batch / seq
        outHead[(((size_t)h * kB + bb) * kS + ss) * kDH + hc] = to_bf16(y);
      } else {
        outF32[(size_t)m * kD + n] = y;
      }
    }
  }
}

// ---------------------------------------------------------------------------
// Causal flash attention. Grid: (S/64, H*B). Block: 128 threads (4 waves),
// each wave owns 16 query rows. Key tiles of 32, online softmax.
// sK stays row-major [key][dh] (K^T fragment == A-pattern); sV is staged
// transposed [dh][key] so PV B-fragments are also two-b128 loads.
// ---------------------------------------------------------------------------
__global__ __launch_bounds__(128) void attn_kernel(
    const bf16* __restrict__ Qh, const bf16* __restrict__ Kh,
    const bf16* __restrict__ Vh, bf16* __restrict__ Ocat) {
  __shared__ __align__(16) bf16 sK [32][kDH + 16];    // [key][dh]
  __shared__ __align__(16) bf16 sVt[kDH][32 + 16];    // [dh][key]
  __shared__ __align__(16) bf16 sP [4][16][32 + 16];  // per-wave P scratch

  const int hb   = blockIdx.y;            // h*B + b
  const int h    = hb / kB;
  const int bb   = hb - h * kB;
  const int m0   = blockIdx.x * 64;       // query tile base
  const int tid  = threadIdx.x;
  const int wave = tid >> 5;
  const int lane = tid & 31;

  const size_t base = (size_t)hb * kS * kDH;
  const bf16* Q = Qh + base;
  const bf16* K = Kh + base;
  const bf16* V = Vh + base;

  // Loop-invariant Q fragments (16 rows x 64 = two K-slices of 32).
  v16bf aq0, aq1;
  {
    const int qrow = m0 + wave * 16 + (lane & 15);
    const int kb8  = (lane >> 4) * 8;
    const bf16* qp = Q + (size_t)qrow * kDH;
    struct alignas(16) P2 { uint4 lo, hi; } t0, t1;
    t0.lo = *reinterpret_cast<const uint4*>(qp + kb8);
    t0.hi = *reinterpret_cast<const uint4*>(qp + kb8 + 16);
    t1.lo = *reinterpret_cast<const uint4*>(qp + 32 + kb8);
    t1.hi = *reinterpret_cast<const uint4*>(qp + 32 + kb8 + 16);
    aq0 = __builtin_bit_cast(v16bf, t0);
    aq1 = __builtin_bit_cast(v16bf, t1);
  }

  float mrow[8], lrow[8];
#pragma unroll
  for (int e = 0; e < 8; ++e) { mrow[e] = -1e30f; lrow[e] = 0.0f; }
  v8f oacc[4] = {};

  const float scale = 0.125f;             // 1/sqrt(64)
  const int ntiles  = m0 / 32 + 2;        // keys [0, m0+64): uniform per block

  for (int t = 0; t < ntiles; ++t) {
    const int kbase = t * 32;
    {
      const int row = tid >> 2;           // key 0..31
      const int seg = (tid & 3) * 16;     // dh segment
      // K tile: row-major, async global -> LDS (2 x 16B).
      const bf16* kp = K + (size_t)(kbase + row) * kDH + seg;
      stage_b128(kp,     &sK[row][seg]);
      stage_b128(kp + 8, &sK[row][seg + 8]);
      // V tile: transpose on store -> sVt[dh][key] (needs the registers).
      struct alignas(16) P2 { uint4 lo, hi; } tv;
      const bf16* vp = V + (size_t)(kbase + row) * kDH + seg;
      tv.lo = *reinterpret_cast<const uint4*>(vp);
      tv.hi = *reinterpret_cast<const uint4*>(vp + 8);
      const B16x16 ve = __builtin_bit_cast(B16x16, tv);
#pragma unroll
      for (int j = 0; j < 16; ++j) sVt[seg + j][row] = ve.v[j];
    }
    stage_join();
    __syncthreads();

    // Scores: Q(16x64) @ K^T -- K^T fragment over row-major sK is A-pattern.
    v8f s0 = {}, s1 = {};
    {
      const v16bf b00 = load_frag<kDH + 16>(&sK[0][0]);
      const v16bf b01 = load_frag<kDH + 16>(&sK[0][32]);
      const v16bf b10 = load_frag<kDH + 16>(&sK[16][0]);
      const v16bf b11 = load_frag<kDH + 16>(&sK[16][32]);
      s0 = wmma_bf16(aq0, b00, s0);
      s0 = wmma_bf16(aq1, b01, s0);
      s1 = wmma_bf16(aq0, b10, s1);
      s1 = wmma_bf16(aq1, b11, s1);
    }

    // Online softmax over this 16x32 score block. Row r lives on a fixed
    // 16-lane half at vector index r&7, so xor-shuffles 1/2/4/8 reduce a row.
    const int colk = lane & 15;
    const int rb   = (lane >> 4) * 8;
    float p0[8], p1[8];
#pragma unroll
    for (int e = 0; e < 8; ++e) {
      const int qg = m0 + wave * 16 + rb + e;
      float v0 = s0[e] * scale;
      float v1 = s1[e] * scale;
      if (kbase + colk > qg)      v0 = -1e30f;   // causal mask
      if (kbase + 16 + colk > qg) v1 = -1e30f;
      float tm = fmaxf(v0, v1);
      tm = fmaxf(tm, __shfl_xor(tm, 1, 32));
      tm = fmaxf(tm, __shfl_xor(tm, 2, 32));
      tm = fmaxf(tm, __shfl_xor(tm, 4, 32));
      tm = fmaxf(tm, __shfl_xor(tm, 8, 32));
      const float mn = fmaxf(mrow[e], tm);
      const float c  = __expf(mrow[e] - mn);
      p0[e] = __expf(v0 - mn);
      p1[e] = __expf(v1 - mn);
      float rs = p0[e] + p1[e];
      rs += __shfl_xor(rs, 1, 32);
      rs += __shfl_xor(rs, 2, 32);
      rs += __shfl_xor(rs, 4, 32);
      rs += __shfl_xor(rs, 8, 32);
      lrow[e] = lrow[e] * c + rs;
      mrow[e] = mn;
#pragma unroll
      for (int n = 0; n < 4; ++n) oacc[n][e] *= c;   // rescale running O
    }

    // Re-stripe P from C-layout to A-layout through LDS.
#pragma unroll
    for (int e = 0; e < 8; ++e) {
      sP[wave][rb + e][colk]      = to_bf16(p0[e]);
      sP[wave][rb + e][16 + colk] = to_bf16(p1[e]);
    }
    __syncthreads();

    // O += P(16x32) @ V(32x64); V B-fragments from transposed sVt.
    {
      const v16bf ap = load_frag<32 + 16>(&sP[wave][0][0]);
      v16bf bv[4];
#pragma unroll
      for (int n = 0; n < 4; ++n)
        bv[n] = load_frag<32 + 16>(&sVt[n * 16][0]);
#pragma unroll
      for (int n = 0; n < 4; ++n)
        oacc[n] = wmma_bf16(ap, bv[n], oacc[n]);
    }
    __syncthreads();
  }

  // Normalize and write head-concat layout [B, S, D] (feature = h*64 + col).
#pragma unroll
  for (int e = 0; e < 8; ++e) {
    const int qg  = m0 + wave * 16 + (lane >> 4) * 8 + e;
    const float inv = 1.0f / lrow[e];
#pragma unroll
    for (int n = 0; n < 4; ++n) {
      const int feat = h * kDH + n * 16 + (lane & 15);
      Ocat[((size_t)bb * kS + qg) * kD + feat] = to_bf16(oacc[n][e] * inv);
    }
  }
}

// ---------------------------------------------------------------------------
extern "C" void kernel_launch(void* const* d_in, const int* in_sizes, int n_in,
                              void* d_out, int out_size, void* d_ws, size_t ws_size,
                              hipStream_t stream) {
  (void)in_sizes; (void)n_in; (void)out_size; (void)ws_size;
  const float* q  = (const float*)d_in[0];
  const float* k  = (const float*)d_in[1];
  const float* v  = (const float*)d_in[2];
  // d_in[3] = causal mask (int) — computed analytically in-kernel.
  const float* Wq = (const float*)d_in[4];
  const float* bq = (const float*)d_in[5];
  const float* Wk = (const float*)d_in[6];
  const float* bk = (const float*)d_in[7];
  const float* Wv = (const float*)d_in[8];
  const float* bv = (const float*)d_in[9];
  const float* Wo = (const float*)d_in[10];
  const float* bo = (const float*)d_in[11];

  char* ws = (char*)d_ws;
  const size_t MB = 1ull << 20;
  bf16* qb  = (bf16*)(ws + 0 * MB);
  bf16* kb  = (bf16*)(ws + 8 * MB);
  bf16* vb  = (bf16*)(ws + 16 * MB);
  bf16* Wqb = (bf16*)(ws + 24 * MB);
  bf16* Wkb = (bf16*)(ws + 26 * MB);
  bf16* Wvb = (bf16*)(ws + 28 * MB);
  bf16* Wob = (bf16*)(ws + 30 * MB);
  bf16* Qh  = (bf16*)(ws + 32 * MB);
  bf16* Kh  = (bf16*)(ws + 40 * MB);
  bf16* Vh  = (bf16*)(ws + 48 * MB);
  bf16* Acat= (bf16*)(ws + 56 * MB);

  const int nXD = kNTok * kD;   // 4M elements
  const int nWW = kD * kD;      // 1M elements
  cvt_f32_bf16_kernel<<<1024, 256, 0, stream>>>(q,  qb,  nXD);
  cvt_f32_bf16_kernel<<<1024, 256, 0, stream>>>(k,  kb,  nXD);
  cvt_f32_bf16_kernel<<<1024, 256, 0, stream>>>(v,  vb,  nXD);
  cvt_f32_bf16_kernel<<<512,  256, 0, stream>>>(Wq, Wqb, nWW);
  cvt_f32_bf16_kernel<<<512,  256, 0, stream>>>(Wk, Wkb, nWW);
  cvt_f32_bf16_kernel<<<512,  256, 0, stream>>>(Wv, Wvb, nWW);
  cvt_f32_bf16_kernel<<<512,  256, 0, stream>>>(Wo, Wob, nWW);

  const dim3 ggrid(kNTok / GBM, kD / GBN);   // 64 x 8 blocks
  gemm_bf16_kernel<<<ggrid, 128, 0, stream>>>(qb, Wqb, bq, Qh, nullptr, 0);
  gemm_bf16_kernel<<<ggrid, 128, 0, stream>>>(kb, Wkb, bk, Kh, nullptr, 0);
  gemm_bf16_kernel<<<ggrid, 128, 0, stream>>>(vb, Wvb, bv, Vh, nullptr, 0);

  attn_kernel<<<dim3(kS / 64, kH * kB), 128, 0, stream>>>(Qh, Kh, Vh, Acat);

  gemm_bf16_kernel<<<ggrid, 128, 0, stream>>>(Acat, Wob, bo, nullptr,
                                              (float*)d_out, 1);
}
